// Recurrence_16449724743872
// MI455X (gfx1250) — matrix operand order repaired and checked by reading, not compiled, not run on hardware
//
#include <hip/hip_runtime.h>
#include <hip/hip_bf16.h>

// ---------------------------------------------------------------------------
// Model constants (from reference)
// ---------------------------------------------------------------------------
#define TT_   32
#define NN_   256
#define NL_   10
#define TE_   128
#define NE_   4
#define CH_   64
#define DD_   16
#define HW_   16
#define INV_  16
#define INVH_ 64
#define HID_  256
#define NA_   16
#define LE_   64
#define Z1IN_ 256   // TE + CH + INVH
#define Z2IN_ 320   // Z1IN + LE
#define KOUT_ 9216  // CH*D*KS*KS
#define OSP_  81    // 9*9 conv output spatial
#define ICK_  144   // D*KS*KS (im2col K)
#define ICKP_ 160   // padded to 32
#define OSPP_ 96    // 81 -> 6 tiles of 16
#define OBSZ_ (DD_*HW_*HW_)   // 4096 per (t,n)

// ---------------------------------------------------------------------------
// WMMA plumbing (CDNA5 / gfx1250, wave32, v_wmma_f32_16x16x32_bf16)
// ---------------------------------------------------------------------------
typedef __attribute__((ext_vector_type(16))) __bf16 v16bf;
typedef __attribute__((ext_vector_type(8)))  __bf16 v8bf;
typedef __attribute__((ext_vector_type(8)))  float  v8f;

// Hardware f32 -> bf16 (clang legalizes fptrunc to the native cvt on gfx1250;
// adjacent casts SLP-fuse into packed converts).
__device__ __forceinline__ __bf16 f2bf(float f) { return static_cast<__bf16>(f); }
__device__ __forceinline__ float sigm(float x) { return 1.f / (1.f + expf(-x)); }

// 16x16 C tile accumulate: acc += A(16 x K) * W^T, bf16 sources (global or LDS).
//   A: row-major bf16, leading dim lda (elements; must be multiple of 16)
//   W: row-major bf16 (outcols x K): B(k, col) = W[col*ldw + k]
// K multiple of 32.  Vectorized b128/b256-class fragment loads.
__device__ __forceinline__ v8f wmma_bf16(const __bf16* __restrict__ A, int lda,
                                         const __bf16* __restrict__ B, int ldb,
                                         int K, v8f acc) {
  const int lane = threadIdx.x & 31;
  const int g = lane >> 4, r = lane & 15;
  const __bf16* ar = A + (long)r * lda;
  const __bf16* bc = B + (long)r * ldb;
  for (int k0 = 0; k0 < K; k0 += 32) {
    v8bf a0 = *(const v8bf*)(ar + k0 + (g << 3));
    v8bf a1 = *(const v8bf*)(ar + k0 + 16 + (g << 3));
    v16bf a = __builtin_shufflevector(a0, a1, 0,1,2,3,4,5,6,7,8,9,10,11,12,13,14,15);
    v8bf b0 = *(const v8bf*)(bc + k0 + (g << 3));
    v8bf b1 = *(const v8bf*)(bc + k0 + 16 + (g << 3));
    v16bf b = __builtin_shufflevector(b0, b1, 0,1,2,3,4,5,6,7,8,9,10,11,12,13,14,15);
    acc = __builtin_amdgcn_wmma_f32_16x16x32_bf16(false, a, false, b,
                                                  (short)0, acc, false, false);
  }
  return acc;
}

// ---------------------------------------------------------------------------
// Phase 0 kernels
// ---------------------------------------------------------------------------

// Generic f32 -> bf16 conversion, 2 elements/thread (one-time staging).
__global__ void k_cvt(const float* __restrict__ src, __bf16* __restrict__ dst, int n) {
  int id = (blockIdx.x * 256 + threadIdx.x) * 2;
  if (id + 1 < n) {
    float2 v = *(const float2*)(src + id);
    dst[id]     = f2bf(v.x);
    dst[id + 1] = f2bf(v.y);                 // adjacent pair -> packed cvt + b32 store
  } else if (id < n) {
    dst[id] = f2bf(src[id]);
  }
}

// M[n, j, :] = mean over 4 tokens of emb_task   -> Mbf (2560 x 128, bf16)
__global__ void k_M(const int* __restrict__ toks, const float* __restrict__ emb,
                    __bf16* __restrict__ Mbf) {
  int id = blockIdx.x * 256 + threadIdx.x;
  if (id >= NN_ * NL_ * TE_) return;
  int row = id / TE_, te = id % TE_;
  float s = 0.f;
#pragma unroll
  for (int tk = 0; tk < 4; ++tk) s += emb[(long)toks[row * 4 + tk] * TE_ + te];
  Mbf[id] = f2bf(s * 0.25f);
}

// Build padded / fused bf16 weight buffers + f32 biases + init pointer state.
__global__ void k_pack(const float* Wb, const float* bb, const float* Winv,
                       const float* Wa, const float* ba, const float* Wc, const float* bc,
                       const float* Wu, const float* bu, const float* Wdg, const float* bdg,
                       const int* p0,
                       __bf16* Wbp, float* bbp, __bf16* Winvp,
                       __bf16* Wh1, float* bh1, __bf16* Wh2, float* bh2, int* p_cur) {
  int tid = threadIdx.x;
  for (int idx = tid; idx < 16 * TE_; idx += 256) {
    int rr = idx / TE_, k = idx % TE_;
    Wbp[idx] = f2bf((rr < NE_) ? Wb[rr * TE_ + k] : 0.f);
  }
  if (tid < 16) bbp[tid] = (tid < NE_) ? bb[tid] : 0.f;
  for (int idx = tid; idx < INVH_ * 32; idx += 256) {
    int rr = idx / 32, k = idx % 32;
    Winvp[idx] = f2bf((k < INV_) ? Winv[rr * INV_ + k] : 0.f);
  }
  for (int idx = tid; idx < 32 * HID_; idx += 256) {
    int rr = idx / HID_, k = idx % HID_;
    float v = 0.f;
    if (rr < NA_) v = Wa[rr * HID_ + k];
    else if (rr == NA_) v = Wc[k];
    Wh1[idx] = f2bf(v);
  }
  if (tid < 32) bh1[tid] = (tid < NA_) ? ba[tid] : (tid == NA_ ? bc[0] : 0.f);
  for (int idx = tid; idx < 16 * Z2IN_; idx += 256) {
    int rr = idx / Z2IN_, k = idx % Z2IN_;
    float v = 0.f;
    if (rr < NE_) v = Wu[rr * Z2IN_ + k];
    else if (rr < 6) v = Wdg[(rr - 4) * Z2IN_ + k];
    Wh2[idx] = f2bf(v);
  }
  if (tid < 16) bh2[tid] = (tid < NE_) ? bu[tid] : (tid < 6 ? bdg[tid - 4] : 0.f);
  if (tid < NN_) p_cur[tid] = p0[tid];
}

// inventory (8192 x 16) -> zero-padded bf16 (8192 x 32)
__global__ void k_pad_inv(const float* __restrict__ inv, __bf16* __restrict__ invp) {
  int id = blockIdx.x * 256 + threadIdx.x;
  if (id >= TT_ * NN_ * 32) return;
  int row = id / 32, c = id % 32;
  invp[id] = f2bf((c < INV_) ? inv[row * INV_ + c] : 0.f);
}

// inv_e_all = relu(inv_pad @ Winv_pad^T + b_inv)  (8192 x 64, K=32) -> bf16
__global__ __launch_bounds__(256) void k_inv_e(const __bf16* __restrict__ invp,
                                               const __bf16* __restrict__ Winvp,
                                               const float* __restrict__ binv,
                                               __bf16* __restrict__ inve) {
  int w = threadIdx.x >> 5, lane = threadIdx.x & 31, g = lane >> 4, r = lane & 15;
  int tile = blockIdx.x * 8 + w;           // 2048 tiles
  int rt = tile >> 2, ct = tile & 3;
  v8f acc = {};
  acc = wmma_bf16(invp + (long)rt * 16 * 32, 32, Winvp + (long)ct * 16 * 32, 32, 32, acc);
  float bias = binv[ct * 16 + r];
#pragma unroll
  for (int d = 0; d < 8; ++d) {
    float v = fmaxf(acc[d] + bias, 0.f);
    inve[(long)(rt * 16 + d + 8 * g) * INVH_ + ct * 16 + r] = f2bf(v);
  }
}

// K_all[(n*NL+j), :] = M @ W_k^T + b_k   (2560 x 9216, K=128) -> bf16 K-cache
__global__ __launch_bounds__(256) void k_Kall(const __bf16* __restrict__ Mbf,
                                              const __bf16* __restrict__ Wkb,
                                              const float* __restrict__ bk,
                                              __bf16* __restrict__ Kall) {
  int w = threadIdx.x >> 5, lane = threadIdx.x & 31, g = lane >> 4, r = lane & 15;
  int ct = blockIdx.x * 8 + w;             // 576 col tiles
  int rt = blockIdx.y;                     // 160 row tiles
  v8f acc = {};
  acc = wmma_bf16(Mbf + (long)rt * 16 * TE_, TE_, Wkb + (long)ct * 16 * TE_, TE_, TE_, acc);
  float bias = bk[ct * 16 + r];
#pragma unroll
  for (int d = 0; d < 8; ++d)
    Kall[(long)(rt * 16 + d + 8 * g) * KOUT_ + ct * 16 + r] = f2bf(acc[d] + bias);
}

// Fused bidirectional-GRU direction + sigmoid(W_beta) head.
// One wave owns 16 batch rows; f32 h + bf16 shadow hb live in LDS across 10 steps.
__global__ __launch_bounds__(64) void k_gru(const __bf16* __restrict__ Mbf,
                                            const __bf16* __restrict__ Wih,
                                            const __bf16* __restrict__ Whh,
                                            const float* __restrict__ bih,
                                            const float* __restrict__ bhh,
                                            const __bf16* __restrict__ Wbp,
                                            const float* __restrict__ bbp,
                                            float* __restrict__ Bsig, int dir) {
  __shared__ __align__(32) float  hS[2][16 * TE_];
  __shared__ __align__(32) __bf16 hbS[2][16 * TE_];
  int w = threadIdx.x >> 5, lane = threadIdx.x & 31, g = lane >> 4, r = lane & 15;
  float*  h  = hS[w];
  __bf16* hb = hbS[w];
  for (int k = lane; k < 16 * TE_; k += 32) { h[k] = 0.f; hb[k] = f2bf(0.f); }
  int b0 = blockIdx.x * 32 + w * 16;       // 2560 rows / 16
  int i = b0 >> 8;                         // roll shift (constant per tile)
  int n0 = b0 & 255;
  for (int s = 0; s < NL_; ++s) {
    int j = dir ? (NL_ - 1 - s) : s;
    int jj = (i + j) % NL_;
    const __bf16* X = Mbf + ((long)n0 * NL_ + jj) * TE_;   // lda = NL_*TE_
    v8f hn8[8];
#pragma unroll
    for (int ct = 0; ct < 8; ++ct) {
      float br = bih[ct * 16 + r] + bhh[ct * 16 + r];
      float bz = bih[TE_ + ct * 16 + r] + bhh[TE_ + ct * 16 + r];
      float bi = bih[2 * TE_ + ct * 16 + r];
      float bh = bhh[2 * TE_ + ct * 16 + r];
      v8f aR = {}, aZ = {}, aI = {}, aH = {};
      aR = wmma_bf16(X,  NL_ * TE_, Wih + (long)(ct * 16) * TE_, TE_, TE_, aR);
      aR = wmma_bf16(hb, TE_,       Whh + (long)(ct * 16) * TE_, TE_, TE_, aR);
      aZ = wmma_bf16(X,  NL_ * TE_, Wih + (long)(TE_ + ct * 16) * TE_, TE_, TE_, aZ);
      aZ = wmma_bf16(hb, TE_,       Whh + (long)(TE_ + ct * 16) * TE_, TE_, TE_, aZ);
      aI = wmma_bf16(X,  NL_ * TE_, Wih + (long)(2 * TE_ + ct * 16) * TE_, TE_, TE_, aI);
      aH = wmma_bf16(hb, TE_,       Whh + (long)(2 * TE_ + ct * 16) * TE_, TE_, TE_, aH);
      v8f hn;
#pragma unroll
      for (int d = 0; d < 8; ++d) {
        float ho = h[(d + 8 * g) * TE_ + ct * 16 + r];
        float rg = sigm(aR[d] + br);
        float zg = sigm(aZ[d] + bz);
        float nn = tanhf(aI[d] + bi + rg * (aH[d] + bh));
        hn[d] = (1.f - zg) * nn + zg * ho;
      }
      hn8[ct] = hn;
    }
#pragma unroll
    for (int ct = 0; ct < 8; ++ct)
#pragma unroll
      for (int d = 0; d < 8; ++d) {
        int idx = (d + 8 * g) * TE_ + ct * 16 + r;
        h[idx]  = hn8[ct][d];
        hb[idx] = f2bf(hn8[ct][d]);       // single conversion per element/step
      }
    // Bsig[i, n, j, dir, :] = sigmoid(h @ W_beta^T + b_beta) (4 cols, padded)
    v8f aB = {};
    aB = wmma_bf16(hb, TE_, Wbp, TE_, TE_, aB);
    if (r < NE_) {
      float bb = bbp[r];
#pragma unroll
      for (int d = 0; d < 8; ++d) {
        int n = n0 + d + 8 * g;
        Bsig[(((long)i * NN_ + n) * NL_ + j) * 8 + dir * 4 + r] = sigm(aB[d] + bb);
      }
    }
  }
}

// Pointer-attention precompute P (10,256,20,4) from Bsig.
__global__ void k_P(const float* __restrict__ Bsig, float* __restrict__ P) {
  int id = blockIdx.x * 256 + threadIdx.x;
  if (id >= NL_ * NN_ * NE_) return;
  int e = id & 3, n = (id >> 2) & 255, i = id >> 10;
  const float* B = Bsig + ((long)i * NN_ + n) * NL_ * 8;
  float B2[20];
#pragma unroll
  for (int j = 0; j < NL_; ++j) {
    B2[2 * j]     = B[j * 8 + e];                  // forward
    B2[2 * j + 1] = B[(NL_ - 1 - j) * 8 + 4 + e];  // backward, reversed
  }
  B2[0] = 0.f;                                     // mask first
  float c = 1.f, Ptmp[20];
#pragma unroll
  for (int d = 0; d < 20; ++d) {
    float zprev = (d == 0) ? 0.f : B2[d - 1];      // zero_last rolled by 1
    c *= (1.f - zprev);
    float bfin = (d == 19) ? 1.f : B2[d];
    Ptmp[d] = bfin * c;
  }
  float* Pp = P + ((long)i * NN_ + n) * 20 * 4;
#pragma unroll
  for (int dd = 0; dd < 20; ++dd) {
    float v = (dd < NL_) ? Ptmp[2 * (NL_ - 1 - dd) + 1] : Ptmp[2 * (dd - NL_)];
    Pp[dd * 4 + e] = v;
  }
}

// ---------------------------------------------------------------------------
// Per-timestep kernels
// ---------------------------------------------------------------------------

// Fill Zb (N x 320, bf16): m (gather by p), inv_e_all[t], emb_lower[L[t]].
__global__ void k_prep(const __bf16* __restrict__ Mbf, const __bf16* __restrict__ inve,
                       const float* __restrict__ embL, const int* __restrict__ L,
                       const int* __restrict__ p_cur, __bf16* __restrict__ Zb, int t) {
  int id = blockIdx.x * 256 + threadIdx.x;
  if (id >= NN_ * Z2IN_) return;
  int n = id / Z2IN_, c = id % Z2IN_;
  __bf16 v;
  if (c < TE_)            v = Mbf[((long)n * NL_ + p_cur[n]) * TE_ + c];
  else if (c < TE_ + CH_) return;                       // conv writes these
  else if (c < Z1IN_)     v = inve[((long)t * NN_ + n) * INVH_ + (c - TE_ - CH_)];
  else                    v = f2bf(embL[(long)L[t * NN_ + n] * LE_ + (c - Z1IN_)]);
  Zb[id] = v;
}

// Per-env dynamic-kernel conv via LDS im2col + WMMA; deterministic spatial sum.
__global__ __launch_bounds__(256) void k_conv(const __bf16* __restrict__ obs_bf,
                                              const __bf16* __restrict__ Kall,
                                              const float* __restrict__ cbias,
                                              const int* __restrict__ p_cur,
                                              __bf16* __restrict__ Zb, int t) {
  __shared__ __align__(32) __bf16 kernS[CH_ * ICKP_];   // 20 KB
  __shared__ __align__(32) __bf16 imS[OSPP_ * ICKP_];   // 30 KB
  __shared__ float convP[24 * 16];
  int n = blockIdx.x, tid = threadIdx.x;
  int p = p_cur[n];
  const __bf16* kern = Kall + ((long)n * NL_ + p) * KOUT_;
  for (int idx = tid; idx < CH_ * ICKP_; idx += 256) {
    int c = idx / ICKP_, k = idx % ICKP_;
    kernS[idx] = (k < ICK_) ? kern[c * ICK_ + k] : f2bf(0.f);
  }
  const __bf16* ob = obs_bf + ((long)t * NN_ + n) * OBSZ_;
  for (int idx = tid; idx < OSPP_ * ICKP_; idx += 256) {
    int col = idx / ICKP_, k = idx % ICKP_;
    __bf16 v = f2bf(0.f);
    if (col < OSP_ && k < ICK_) {
      int d = k / 9, rem = k % 9, ky = rem / 3, kx = rem % 3;
      int oy = col / 9, ox = col % 9;
      int iy = oy * 2 + ky - 2, ix = ox * 2 + kx - 2;
      if (iy >= 0 && iy < HW_ && ix >= 0 && ix < HW_) v = ob[d * 256 + iy * 16 + ix];
    }
    imS[idx] = v;
  }
  __syncthreads();
  int w = tid >> 5, lane = tid & 31, g = lane >> 4, r = lane & 15;
  for (int tt = w; tt < 24; tt += 8) {                  // 4 x 6 tiles, uniform per wave
    int mt = tt / 6, nt = tt % 6;
    v8f acc;
#pragma unroll
    for (int d = 0; d < 8; ++d) acc[d] = cbias[mt * 16 + d + 8 * g];
    acc = wmma_bf16(kernS + mt * 16 * ICKP_, ICKP_, imS + nt * 16 * ICKP_, ICKP_, ICKP_, acc);
    int col = nt * 16 + r;
#pragma unroll
    for (int d = 0; d < 8; ++d) {
      float v = fmaxf(acc[d], 0.f);
      v = (col < OSP_) ? v : 0.f;                       // relu + spatial mask
      v += __shfl_xor(v, 1); v += __shfl_xor(v, 2);
      v += __shfl_xor(v, 4); v += __shfl_xor(v, 8);     // sum over 16 cols
      if (r == 0) convP[tt * 16 + d + 8 * g] = v;
    }
  }
  __syncthreads();
  if (tid < CH_) {
    float s = 0.f;
    int mt = tid >> 4, cl = tid & 15;
#pragma unroll
    for (int nt = 0; nt < 6; ++nt) s += convP[(mt * 6 + nt) * 16 + cl];
    Zb[(long)n * Z2IN_ + TE_ + tid] = f2bf(s);
  }
  if (t < TT_ - 1)                                      // warm L2 for next step's obs
    __builtin_prefetch(obs_bf + ((long)(t + 1) * NN_ + n) * OBSZ_ + tid * 16, 0, 1);
}

// z1 = relu(Zb[:, :256] @ W_z1^T + b_z1) -> bf16
__global__ __launch_bounds__(256) void k_z1(const __bf16* __restrict__ Zb,
                                            const __bf16* __restrict__ Wz1,
                                            const float* __restrict__ bz1,
                                            __bf16* __restrict__ z1b) {
  int w = threadIdx.x >> 5, lane = threadIdx.x & 31, g = lane >> 4, r = lane & 15;
  int tile = blockIdx.x * 8 + w;           // 256 tiles
  int rt = tile >> 4, ct = tile & 15;
  v8f acc = {};
  acc = wmma_bf16(Zb + (long)rt * 16 * Z2IN_, Z2IN_, Wz1 + (long)ct * 16 * HID_, HID_, HID_, acc);
  float bias = bz1[ct * 16 + r];
#pragma unroll
  for (int d = 0; d < 8; ++d) {
    float v = fmaxf(acc[d] + bias, 0.f);
    z1b[(long)(rt * 16 + d + 8 * g) * HID_ + ct * 16 + r] = f2bf(v);
  }
}

// heads: H1 = z1 @ Wh1^T (32 cols: a|v, K=256); H2 = Zb @ Wh2^T (16 cols: u|dg, K=320)
__global__ __launch_bounds__(256) void k_heads(const __bf16* __restrict__ z1b,
                                               const __bf16* __restrict__ Zb,
                                               const __bf16* __restrict__ Wh1,
                                               const float* __restrict__ bh1,
                                               const __bf16* __restrict__ Wh2,
                                               const float* __restrict__ bh2,
                                               float* __restrict__ heads) {
  int w = threadIdx.x >> 5, lane = threadIdx.x & 31, g = lane >> 4, r = lane & 15;
  int tile = blockIdx.x * 8 + w;           // 48 tiles (uniform branch per wave)
  if (tile < 32) {
    int rt = tile >> 1, ct = tile & 1;
    v8f acc = {};
    acc = wmma_bf16(z1b + (long)rt * 16 * HID_, HID_, Wh1 + (long)ct * 16 * HID_, HID_, HID_, acc);
    float bias = bh1[ct * 16 + r];
#pragma unroll
    for (int d = 0; d < 8; ++d)
      heads[(long)(rt * 16 + d + 8 * g) * 48 + ct * 16 + r] = acc[d] + bias;
  } else {
    int rt = tile - 32;
    v8f acc = {};
    acc = wmma_bf16(Zb + (long)rt * 16 * Z2IN_, Z2IN_, Wh2, Z2IN_, Z2IN_, acc);
    float bias = bh2[r];
#pragma unroll
    for (int d = 0; d < 8; ++d)
      heads[(long)(rt * 16 + d + 8 * g) * 48 + 32 + r] = acc[d] + bias;
  }
}

// Softmaxes, pointer mixture, argmax, pointer update, output write.
__global__ __launch_bounds__(256) void k_final(const float* __restrict__ heads,
                                               const float* __restrict__ P,
                                               int* __restrict__ p_cur,
                                               float* __restrict__ out, int t) {
  int n = threadIdx.x;
  const float* h = heads + (long)n * 48;
  float* o = out + ((long)t * NN_ + n) * 37;
  float m = h[0];
#pragma unroll
  for (int j = 1; j < NA_; ++j) m = fmaxf(m, h[j]);
  float a[NA_], s = 0.f;
#pragma unroll
  for (int j = 0; j < NA_; ++j) { a[j] = expf(h[j] - m); s += a[j]; }
#pragma unroll
  for (int j = 0; j < NA_; ++j) o[j] = a[j] / s;
  float v = h[16];
  float mu = fmaxf(fmaxf(h[32], h[33]), fmaxf(h[34], h[35]));
  float u[NE_], su = 0.f;
#pragma unroll
  for (int e = 0; e < NE_; ++e) { u[e] = expf(h[32 + e] - mu); su += u[e]; }
#pragma unroll
  for (int e = 0; e < NE_; ++e) u[e] /= su;
  bool gg = h[37] > h[36];                 // argmax(dg) == 1 (ties -> 0)
  int p = p_cur[n];
  const float* Pp = P + ((long)p * NN_ + n) * 20 * 4;
  float dp[20];
#pragma unroll
  for (int d = 0; d < 20; ++d) {
    float val;
    if (gg) {
      val = 0.f;
#pragma unroll
      for (int e = 0; e < NE_; ++e) val += Pp[d * 4 + e] * u[e];
    } else {
      val = (d == NL_) ? 1.f : 0.f;        // one-hot "stay"
    }
    dp[d] = val;
    o[16 + d] = val;
  }
  o[36] = v;
  int dnew = 0; float best = dp[0];
#pragma unroll
  for (int d = 1; d < 20; ++d) if (dp[d] > best) { best = dp[d]; dnew = d; }
  int pn = p + dnew - NL_;
  pn = pn < 0 ? 0 : (pn > NL_ - 1 ? NL_ - 1 : pn);
  p_cur[n] = pn;
}

// ---------------------------------------------------------------------------
// Host launcher.  Workspace requirement: ~125 MB.
// ---------------------------------------------------------------------------
extern "C" void kernel_launch(void* const* d_in, const int* in_sizes, int n_in,
                              void* d_out, int out_size, void* d_ws, size_t ws_size,
                              hipStream_t stream) {
  const float* obs       = (const float*)d_in[0];
  const float* inventory = (const float*)d_in[1];
  const int*   lines_tok = (const int*)d_in[2];
  const int*   L         = (const int*)d_in[3];
  const int*   p0        = (const int*)d_in[4];
  const float* emb_task  = (const float*)d_in[5];
  const float* wih_f = (const float*)d_in[6];
  const float* whh_f = (const float*)d_in[7];
  const float* bih_f = (const float*)d_in[8];
  const float* bhh_f = (const float*)d_in[9];
  const float* wih_b = (const float*)d_in[10];
  const float* whh_b = (const float*)d_in[11];
  const float* bih_b = (const float*)d_in[12];
  const float* bhh_b = (const float*)d_in[13];
  const float* W_beta = (const float*)d_in[14];
  const float* b_beta = (const float*)d_in[15];
  const float* W_k    = (const float*)d_in[16];
  const float* b_k    = (const float*)d_in[17];
  const float* cbias  = (const float*)d_in[18];
  const float* W_inv  = (const float*)d_in[19];
  const float* b_inv  = (const float*)d_in[20];
  const float* W_z1   = (const float*)d_in[21];
  const float* b_z1   = (const float*)d_in[22];
  const float* W_a    = (const float*)d_in[23];
  const float* b_a    = (const float*)d_in[24];
  const float* embL   = (const float*)d_in[25];
  const float* W_u    = (const float*)d_in[26];
  const float* b_u    = (const float*)d_in[27];
  const float* W_dg   = (const float*)d_in[28];
  const float* b_dg   = (const float*)d_in[29];
  const float* W_c    = (const float*)d_in[30];
  const float* b_c    = (const float*)d_in[31];
  float* out = (float*)d_out;

  char* ws = (char*)d_ws;
  size_t off = 0;
  auto alloc = [&](size_t bytes) -> void* {
    void* p = ws + off;
    off = (off + bytes + 255) & ~(size_t)255;
    return p;
  };
  __bf16* Mbf    = (__bf16*)alloc((size_t)NN_ * NL_ * TE_ * 2);
  __bf16* Wkb    = (__bf16*)alloc((size_t)KOUT_ * TE_ * 2);
  __bf16* Kall   = (__bf16*)alloc((size_t)NN_ * NL_ * KOUT_ * 2);   // 47 MB
  __bf16* obs_bf = (__bf16*)alloc((size_t)TT_ * NN_ * OBSZ_ * 2);   // 67 MB
  __bf16* wihfb  = (__bf16*)alloc((size_t)3 * TE_ * TE_ * 2);
  __bf16* whhfb  = (__bf16*)alloc((size_t)3 * TE_ * TE_ * 2);
  __bf16* wihbb  = (__bf16*)alloc((size_t)3 * TE_ * TE_ * 2);
  __bf16* whhbb  = (__bf16*)alloc((size_t)3 * TE_ * TE_ * 2);
  __bf16* Wbpb   = (__bf16*)alloc(16 * TE_ * 2);
  float*  bbp    = (float*)alloc(16 * 4);
  __bf16* Winvpb = (__bf16*)alloc(INVH_ * 32 * 2);
  __bf16* Wh1b   = (__bf16*)alloc(32 * HID_ * 2);
  float*  bh1    = (float*)alloc(32 * 4);
  __bf16* Wh2b   = (__bf16*)alloc(16 * Z2IN_ * 2);
  float*  bh2    = (float*)alloc(16 * 4);
  __bf16* invpb  = (__bf16*)alloc((size_t)TT_ * NN_ * 32 * 2);
  __bf16* inve   = (__bf16*)alloc((size_t)TT_ * NN_ * INVH_ * 2);
  float*  Bsig   = (float*)alloc((size_t)NL_ * NN_ * NL_ * 8 * 4);
  float*  Pws    = (float*)alloc((size_t)NL_ * NN_ * 20 * 4 * 4);
  __bf16* Zb     = (__bf16*)alloc((size_t)NN_ * Z2IN_ * 2);
  __bf16* z1b    = (__bf16*)alloc((size_t)NN_ * HID_ * 2);
  float*  hds    = (float*)alloc((size_t)NN_ * 48 * 4);
  int*    p_cur  = (int*)alloc(NN_ * 4);
  __bf16* Wz1b   = (__bf16*)alloc((size_t)HID_ * HID_ * 2);

  auto cvt_grid = [](int n) { return (n / 2 + 255) / 256; };

  // ---- phase 0: one-time bf16 staging + precomputes ----
  k_M<<<(NN_ * NL_ * TE_ + 255) / 256, 256, 0, stream>>>(lines_tok, emb_task, Mbf);
  k_cvt<<<cvt_grid(KOUT_ * TE_), 256, 0, stream>>>(W_k, Wkb, KOUT_ * TE_);
  k_cvt<<<cvt_grid(TT_ * NN_ * OBSZ_), 256, 0, stream>>>(obs, obs_bf, TT_ * NN_ * OBSZ_);
  k_cvt<<<cvt_grid(3 * TE_ * TE_), 256, 0, stream>>>(wih_f, wihfb, 3 * TE_ * TE_);
  k_cvt<<<cvt_grid(3 * TE_ * TE_), 256, 0, stream>>>(whh_f, whhfb, 3 * TE_ * TE_);
  k_cvt<<<cvt_grid(3 * TE_ * TE_), 256, 0, stream>>>(wih_b, wihbb, 3 * TE_ * TE_);
  k_cvt<<<cvt_grid(3 * TE_ * TE_), 256, 0, stream>>>(whh_b, whhbb, 3 * TE_ * TE_);
  k_cvt<<<cvt_grid(HID_ * HID_), 256, 0, stream>>>(W_z1, Wz1b, HID_ * HID_);
  k_pack<<<1, 256, 0, stream>>>(W_beta, b_beta, W_inv, W_a, b_a, W_c, b_c,
                                W_u, b_u, W_dg, b_dg, p0,
                                Wbpb, bbp, Winvpb, Wh1b, bh1, Wh2b, bh2, p_cur);
  k_pad_inv<<<(TT_ * NN_ * 32 + 255) / 256, 256, 0, stream>>>(inventory, invpb);
  k_inv_e<<<256, 256, 0, stream>>>(invpb, Winvpb, b_inv, inve);
  k_Kall<<<dim3(72, 160), 256, 0, stream>>>(Mbf, Wkb, b_k, Kall);
  k_gru<<<80, 64, 0, stream>>>(Mbf, wihfb, whhfb, bih_f, bhh_f, Wbpb, bbp, Bsig, 0);
  k_gru<<<80, 64, 0, stream>>>(Mbf, wihbb, whhbb, bih_b, bhh_b, Wbpb, bbp, Bsig, 1);
  k_P<<<(NL_ * NN_ * NE_ + 255) / 256, 256, 0, stream>>>(Bsig, Pws);

  // ---- serial pointer recurrence over T steps (stream-ordered) ----
  for (int t = 0; t < TT_; ++t) {
    k_prep<<<(NN_ * Z2IN_ + 255) / 256, 256, 0, stream>>>(Mbf, inve, embL, L, p_cur, Zb, t);
    k_conv<<<NN_, 256, 0, stream>>>(obs_bf, Kall, cbias, p_cur, Zb, t);
    k_z1<<<32, 256, 0, stream>>>(Zb, Wz1b, b_z1, z1b);
    k_heads<<<6, 256, 0, stream>>>(z1b, Zb, Wh1b, bh1, Wh2b, bh2, hds);
    k_final<<<1, 256, 0, stream>>>(hds, Pws, p_cur, out, t);
  }
}
